// IncrSeqAttnMatch_60911226192260
// MI455X (gfx1250) — matrix operand order repaired and checked by reading, not compiled, not run on hardware
//
#include <hip/hip_runtime.h>
#include <math.h>

// ---------------------------------------------------------------------------
// IncrSeqAttnMatch on MI455X (gfx1250, wave32, WMMA).
// All three matmuls (projection, Q·K^T, alpha·V) use v_wmma_f32_16x16x32_bf16
// with fp32 accumulation; softmax is fp32. Keys are kept in bf16 in BOTH
// row-major [key,h] and transposed [h,key] layouts so every WMMA A/B fragment
// is a contiguous 16B/32B load matching the CDNA5 wave32 VGPR layouts.
// The alpha·V GEMM is split-K across 8 blocks (fp32 partials + deterministic
// fixed-order combine) to keep the per-step critical path wide.
// ---------------------------------------------------------------------------

typedef __bf16 bf16;
typedef __attribute__((ext_vector_type(16))) bf16  v16bf;
typedef __attribute__((ext_vector_type(8)))  bf16  v8bf;
typedef __attribute__((ext_vector_type(8)))  float v8f;

constexpr int Hd    = 512;          // hidden
constexpr int LQ    = 256;          // query tokens per turn
constexpr int LA    = 256;          // answer tokens per turn
constexpr int LL    = 512;          // LQ + LA
constexpr int NB    = 128;          // turns (B)
constexpr int NK    = NB * LL;      // 65536 total key rows
constexpr int LDK   = 65536;        // row stride (elements) for S and P
constexpr int SPLIT = 8;            // split-K factor for alpha*V

// --- WMMA fragment loaders (CDNA5 wave32 layouts, CDNA5 ISA §7.12.2) -------
// A (16x32 bf16): lane&15 = M row; lanes<16 hold K {k0..k0+7, k0+16..k0+23},
// lanes>=16 hold K {k0+8..k0+15, k0+24..k0+31}  -> two 16B contiguous loads.
__device__ __forceinline__ v16bf ldA(const bf16* __restrict__ rowptr, int k0, int lane) {
  int hl = lane >> 4;
  union { v16bf v; v8bf h[2]; } u;
  u.h[0] = *(const v8bf*)(rowptr + k0 + hl * 8);
  u.h[1] = *(const v8bf*)(rowptr + k0 + 16 + hl * 8);
  return u.v;
}
// A from fp32 source (projection inputs), converted to bf16 on the fly.
__device__ __forceinline__ v16bf ldA_f32(const float* __restrict__ rowptr, int k0, int lane) {
  int hl = lane >> 4;
  v8f f0 = *(const v8f*)(rowptr + k0 + hl * 8);
  v8f f1 = *(const v8f*)(rowptr + k0 + 16 + hl * 8);
  union { v16bf v; v8bf h[2]; } u;
#pragma unroll
  for (int i = 0; i < 8; i++) { u.h[0][i] = (bf16)f0[i]; u.h[1][i] = (bf16)f1[i]; }
  return u.v;
}
// B (32x16 bf16): lane&15 = N col; lanes<16 hold K {k0..k0+7, k0+16..k0+23},
// lanes>=16 hold K {k0+8..k0+15, k0+24..k0+31} -> two 16B contiguous loads.
// colptr must point at &B[col * ld] where the column's K data is contiguous.
__device__ __forceinline__ v16bf ldB(const bf16* __restrict__ colptr, int k0, int lane) {
  int hl = lane >> 4;
  union { v16bf v; v8bf h[2]; } u;
  u.h[0] = *(const v8bf*)(colptr + k0 + hl * 8);
  u.h[1] = *(const v8bf*)(colptr + k0 + 16 + hl * 8);
  return u.v;
}

__device__ __forceinline__ v8f wmma_bf16(v16bf a, v16bf b, v8f c) {
  return __builtin_amdgcn_wmma_f32_16x16x32_bf16(false, a, false, b, (short)0, c, false, false);
}

// --- W fp32 -> bf16 ---------------------------------------------------------
__global__ void convW_kernel(const float* __restrict__ W, bf16* __restrict__ Wbf) {
  int i = blockIdx.x * blockDim.x + threadIdx.x;
  if (i < Hd * Hd) Wbf[i] = (bf16)W[i];
}

// --- Projection: relu(X W^T + b) for both streams ---------------------------
// out[g,k] = relu(sum_h X[g,h] * W[k,h] + b[k]); writes Kbuf, KbufT (bf16)
// and, for xq rows, the fp32 output buffer (later overwritten for t>=1).
__global__ void proj_kernel(const float* __restrict__ xq, const float* __restrict__ xa,
                            const float* __restrict__ bias, const bf16* __restrict__ Wbf,
                            bf16* __restrict__ Kbuf, bf16* __restrict__ KbufT,
                            float* __restrict__ out) {
  int wave = (blockIdx.x * blockDim.x + threadIdx.x) >> 5;
  int lane = threadIdx.x & 31;
  int ntile = wave & 15;          // 16 tiles of 32 output features
  int mtile = wave >> 4;          // 4096 tiles of 16 key rows
  if (mtile >= NK / 16) return;
  int g0 = mtile * 16;
  int n0 = ntile * 32;
  int laneN = lane & 15, hl = lane >> 4;

  int g = g0 + laneN;             // key row this lane's A data comes from
  int b = g >> 9;                 // turn
  int l = g & 511;                // position within turn (tiles never straddle)
  const float* arow = (l < LQ) ? (xq + ((size_t)b * LQ + l) * Hd)
                               : (xa + ((size_t)b * LA + (l - LQ)) * Hd);
  const bf16* bcol0 = Wbf + (size_t)(n0 + laneN) * Hd;       // W row k = B column
  const bf16* bcol1 = Wbf + (size_t)(n0 + 16 + laneN) * Hd;

  v8f c0 = {}; v8f c1 = {};
  for (int h0 = 0; h0 < Hd; h0 += 32) {
    v16bf a  = ldA_f32(arow, h0, lane);
    v16bf b0 = ldB(bcol0, h0, lane);
    v16bf b1 = ldB(bcol1, h0, lane);
    c0 = wmma_bf16(a, b0, c0);
    c1 = wmma_bf16(a, b1, c1);
  }
#pragma unroll
  for (int f = 0; f < 2; f++) {
    v8f c = f ? c1 : c0;
    int k = n0 + f * 16 + laneN;
    float bv = bias[k];
#pragma unroll
    for (int r = 0; r < 8; r++) {
      int row = g0 + r + hl * 8;             // C/D layout: vgpr r -> row m0+r(+8)
      float v = c[r] + bv;
      v = v > 0.f ? v : 0.f;
      bf16 vb = (bf16)v;
      Kbuf[(size_t)row * Hd + k]  = vb;
      KbufT[(size_t)k * NK + row] = vb;
      int bl = row & 511;
      if (bl < LQ) out[((size_t)(row >> 9) * LQ + bl) * Hd + k] = v;
    }
  }
}

// --- Step t: S = Q K^T over the t*512-key history (16x32 tiles) -------------
__global__ void scores_kernel(const bf16* __restrict__ Kbuf, float* __restrict__ S, int t) {
  int lane = threadIdx.x & 31;    // blockDim = 32 (one wave per 16x32 tile)
  int j0 = blockIdx.x * 32;       // key tile (blockIdx.x < t*16)
  int m0 = blockIdx.y * 16;       // query tile (16 tiles)
  int laneN = lane & 15, hl = lane >> 4;
  const bf16* qrow  = Kbuf + ((size_t)t * LL + m0 + laneN) * Hd; // original xq_proj[t]
  const bf16* kcol0 = Kbuf + (size_t)(j0 + laneN) * Hd;          // key j, contiguous in h
  const bf16* kcol1 = Kbuf + (size_t)(j0 + 16 + laneN) * Hd;
  v8f c0 = {}, c1 = {};
  for (int h0 = 0; h0 < Hd; h0 += 32) {
    v16bf a  = ldA(qrow, h0, lane);
    v16bf b0 = ldB(kcol0, h0, lane);
    v16bf b1 = ldB(kcol1, h0, lane);
    c0 = wmma_bf16(a, b0, c0);
    c1 = wmma_bf16(a, b1, c1);
  }
#pragma unroll
  for (int f = 0; f < 2; f++) {
    v8f c = f ? c1 : c0;
#pragma unroll
    for (int r = 0; r < 8; r++)
      S[(size_t)(m0 + r + hl * 8) * LDK + j0 + f * 16 + laneN] = c[r];
  }
}

// --- online (max,sum) merge for softmax -------------------------------------
__device__ __forceinline__ void msum_merge(float& m1, float& l1, float m2, float l2) {
  float nm = fmaxf(m1, m2);
  float e1 = (m1 == -INFINITY) ? 0.f : __expf(m1 - nm);
  float e2 = (m2 == -INFINITY) ? 0.f : __expf(m2 - nm);
  l1 = l1 * e1 + l2 * e2;
  m1 = nm;
}

// --- Step t: row softmax over t*512 keys, pad-mask applied, P -> bf16 -------
__global__ void softmax_kernel(const float* __restrict__ S, bf16* __restrict__ P,
                               const unsigned char* __restrict__ qmask,
                               const unsigned char* __restrict__ amask, int t) {
  int m = blockIdx.x;             // 256 query rows
  int tid = threadIdx.x;          // 256 threads
  int N = t * LL;
  const float* srow = S + (size_t)m * LDK;

  float mx = -INFINITY, sum = 0.f;
  for (int j = tid; j < N; j += 256) {
    int jt = j >> 9, jl = j & 511;
    bool pad = (jl < LQ) ? (qmask[jt * LQ + jl] != 0) : (amask[jt * LA + (jl - LQ)] != 0);
    if (pad) continue;
    msum_merge(mx, sum, srow[j], 1.f);
  }
  // wave32 reduction
  for (int off = 16; off > 0; off >>= 1) {
    float omx = __shfl_xor(mx, off, 32);
    float osm = __shfl_xor(sum, off, 32);
    msum_merge(mx, sum, omx, osm);
  }
  __shared__ float smx[8], ssum[8];
  int w = tid >> 5;
  if ((tid & 31) == 0) { smx[w] = mx; ssum[w] = sum; }
  __syncthreads();
  float gmx = smx[0], gsum = ssum[0];
  for (int i = 1; i < 8; i++) msum_merge(gmx, gsum, smx[i], ssum[i]);
  float inv = (gsum > 0.f) ? 1.f / gsum : 0.f;

  for (int j = tid; j < N; j += 256) {
    int jt = j >> 9, jl = j & 511;
    bool pad = (jl < LQ) ? (qmask[jt * LQ + jl] != 0) : (amask[jt * LA + (jl - LQ)] != 0);
    float p = (!pad && gsum > 0.f) ? __expf(srow[j] - gmx) * inv : 0.f;
    P[(size_t)m * LDK + j] = (bf16)p;
  }
}

// --- Step t: split-K partial of hist = P * K --------------------------------
// Grid: (16 n-tiles of 32, 16 m-tiles, SPLIT). Each block owns a disjoint,
// contiguous slice of the t*512 K range and writes fp32 partials.
__global__ void pv_partial_kernel(const bf16* __restrict__ P, const bf16* __restrict__ KbufT,
                                  float* __restrict__ Hpart, int t) {
  int lane = threadIdx.x & 31;    // blockDim = 32
  int n0 = blockIdx.x * 32;       // 16 tiles of 32 output features
  int m0 = blockIdx.y * 16;       // 16 query tiles
  int s  = blockIdx.z;            // split-K slice
  int laneN = lane & 15, hl = lane >> 4;
  int slice = (t * LL) / SPLIT;   // multiple of 64 for t>=1
  int jbeg = s * slice, jend = jbeg + slice;

  const bf16* arow  = P + (size_t)(m0 + laneN) * LDK;
  const bf16* bcol0 = KbufT + (size_t)(n0 + laneN) * NK;        // contiguous along key
  const bf16* bcol1 = KbufT + (size_t)(n0 + 16 + laneN) * NK;
  v8f c0 = {}, c1 = {};
  for (int j0 = jbeg; j0 < jend; j0 += 32) {
    v16bf a  = ldA(arow, j0, lane);
    v16bf b0 = ldB(bcol0, j0, lane);
    v16bf b1 = ldB(bcol1, j0, lane);
    c0 = wmma_bf16(a, b0, c0);
    c1 = wmma_bf16(a, b1, c1);
  }
#pragma unroll
  for (int f = 0; f < 2; f++) {
    v8f c = f ? c1 : c0;
    int k = n0 + f * 16 + laneN;
#pragma unroll
    for (int r = 0; r < 8; r++) {
      int m = m0 + r + hl * 8;
      Hpart[((size_t)s * LQ + m) * Hd + k] = c[r];
    }
  }
}

// --- Step t: fixed-order combine of split-K partials + merge + write-back ---
__global__ void combine_kernel(const float* __restrict__ Hpart,
                               bf16* __restrict__ Kbuf, bf16* __restrict__ KbufT,
                               float* __restrict__ out, int t) {
  int idx = blockIdx.x * blockDim.x + threadIdx.x;   // over 256*512
  if (idx >= LQ * Hd) return;
  int m = idx >> 9;
  int k = idx & 511;
  float h = 0.f;
#pragma unroll
  for (int s = 0; s < SPLIT; s++) h += Hpart[((size_t)s * LQ + m) * Hd + k];
  size_t orow = ((size_t)t * LQ + m) * Hd + k;
  float v = 0.5f * (out[orow] + h);    // out still holds xq_proj[t] here
  out[orow] = v;
  bf16 vb = (bf16)v;
  Kbuf[((size_t)t * LL + m) * Hd + k] = vb;           // history seen by later steps
  KbufT[(size_t)k * NK + (size_t)t * LL + m] = vb;
}

extern "C" void kernel_launch(void* const* d_in, const int* in_sizes, int n_in,
                              void* d_out, int out_size, void* d_ws, size_t ws_size,
                              hipStream_t stream) {
  const float* xq = (const float*)d_in[0];
  const float* xa = (const float*)d_in[1];
  const unsigned char* qmask = (const unsigned char*)d_in[2];
  const unsigned char* amask = (const unsigned char*)d_in[3];
  const float* W    = (const float*)d_in[4];
  const float* bias = (const float*)d_in[5];
  float* out = (float*)d_out;

  // workspace layout (~240 MB): Kbuf | KbufT | S | P | Wbf | Hpart
  char* ws = (char*)d_ws;
  bf16*  Kbuf  = (bf16*)(ws);                                // 64 MB
  bf16*  KbufT = (bf16*)(ws + (size_t)67108864);             // 64 MB
  float* S     = (float*)(ws + (size_t)134217728);           // 64 MB
  bf16*  P     = (bf16*)(ws + (size_t)201326592);            // 32 MB
  bf16*  Wbf   = (bf16*)(ws + (size_t)234881024);            // 0.5 MB
  float* Hpart = (float*)(ws + (size_t)235405312);           // 4 MB

  convW_kernel<<<(Hd * Hd + 255) / 256, 256, 0, stream>>>(W, Wbf);
  proj_kernel<<<8192, 256, 0, stream>>>(xq, xa, bias, Wbf, Kbuf, KbufT, out);

  for (int t = 1; t < NB; t++) {
    scores_kernel<<<dim3(t * 16, 16), 32, 0, stream>>>(Kbuf, S, t);
    softmax_kernel<<<256, 256, 0, stream>>>(S, P, qmask, amask, t);
    pv_partial_kernel<<<dim3(16, 16, SPLIT), 32, 0, stream>>>(P, KbufT, Hpart, t);
    combine_kernel<<<(LQ * Hd + 255) / 256, 256, 0, stream>>>(Hpart, Kbuf, KbufT, out, t);
  }
}
